// MultiFlashAttentionRoPE_38259568672848
// MI455X (gfx1250) — matrix operand order repaired and checked by reading, not compiled, not run on hardware
//
#include <hip/hip_runtime.h>

typedef __attribute__((ext_vector_type(16))) __bf16 v16bf;
typedef __attribute__((ext_vector_type(8)))  __bf16 v8bf;
typedef __attribute__((ext_vector_type(8)))  float  v8f;

#define B_DIM 4
#define S_DIM 2048
#define E_DIM 2048
#define H_DIM 16
#define D_DIM 128

static __device__ __forceinline__ unsigned short f2bf_bits(float f) {
  return __builtin_bit_cast(unsigned short, (__bf16)f);
}

static __device__ __forceinline__ v16bf join16(v8bf lo, v8bf hi) {
  return __builtin_shufflevector(lo, hi, 0,1,2,3,4,5,6,7,8,9,10,11,12,13,14,15);
}

// CDNA5 async DMA: 16B global -> LDS, tracked by ASYNCcnt (per-lane addresses).
static __device__ __forceinline__ void async_ld16(const __bf16* gptr, const __bf16* lptr) {
  unsigned lds_off = (unsigned)(size_t)lptr;                 // low 32b of generic ptr = LDS offset
  unsigned long long ga = (unsigned long long)(size_t)gptr;
  asm volatile("global_load_async_to_lds_b128 %0, %1, off"
               :: "v"(lds_off), "v"(ga) : "memory");
}

// ---------------------------------------------------------------------------
// fp32 -> bf16 conversion (8-byte packed stores)
// ---------------------------------------------------------------------------
__global__ __launch_bounds__(256) void cvt_f32_bf16(const float* __restrict__ src,
                                                    __bf16* __restrict__ dst, int n4) {
  int i = blockIdx.x * blockDim.x + threadIdx.x;
  if (i < n4) {
    float4 v = ((const float4*)src)[i];
    unsigned long long p =
        (unsigned long long)f2bf_bits(v.x) |
        ((unsigned long long)f2bf_bits(v.y) << 16) |
        ((unsigned long long)f2bf_bits(v.z) << 32) |
        ((unsigned long long)f2bf_bits(v.w) << 48);
    ((unsigned long long*)dst)[i] = p;
  }
}

// ---------------------------------------------------------------------------
// QKV projection GEMM: Y = X @ W^T + b  (M=8192, N=2048, K=2048), bf16 WMMA.
// blockIdx.z: 0 -> Q (layout [B,H,S,D]), 1 -> K ([B,H,S,D]), 2 -> V^T ([B,H,D,S])
// wave tile 64x64, workgroup tile 128x256 (8 waves, 2m x 4n)
// ---------------------------------------------------------------------------
__global__ __launch_bounds__(256) void qkv_gemm(
    const __bf16* __restrict__ xq, const __bf16* __restrict__ xkv,
    const __bf16* __restrict__ wqm, const __bf16* __restrict__ wkm, const __bf16* __restrict__ wvm,
    const float* __restrict__ bq, const float* __restrict__ bk, const float* __restrict__ bv,
    __bf16* __restrict__ qout, __bf16* __restrict__ kout, __bf16* __restrict__ vtout) {
  const int z = blockIdx.z;
  const __bf16* X = (z == 0) ? xq : xkv;
  const __bf16* W = (z == 0) ? wqm : ((z == 1) ? wkm : wvm);
  const float* bias = (z == 0) ? bq : ((z == 1) ? bk : bv);

  const int lane = threadIdx.x & 31;
  const int wave = threadIdx.x >> 5;
  const int m0 = blockIdx.x * 128 + (wave & 1) * 64;
  const int n0 = blockIdx.y * 256 + (wave >> 1) * 64;
  const int l15 = lane & 15;
  const int grp = (lane >= 16) ? 1 : 0;
  const int kb8 = grp * 8;    // unused; A-frag also uses kbb
  const int kbb = grp * 16;   // B-frag K offset per lane group

  v8f acc[4][4];
#pragma unroll
  for (int mi = 0; mi < 4; ++mi)
#pragma unroll
    for (int ni = 0; ni < 4; ++ni) acc[mi][ni] = {};

  const __bf16* Arow[4];
  const __bf16* Brow[4];
#pragma unroll
  for (int i = 0; i < 4; ++i) {
    Arow[i] = X + (size_t)(m0 + i * 16 + l15) * E_DIM;
    Brow[i] = W + (size_t)(n0 + i * 16 + l15) * E_DIM;
  }

#pragma unroll 2
  for (int kk = 0; kk < E_DIM; kk += 32) {
    v16bf a[4];
#pragma unroll
    for (int mi = 0; mi < 4; ++mi) {
      v8bf lo = *(const v8bf*)(Arow[mi] + kk + kbb);
      v8bf hi = *(const v8bf*)(Arow[mi] + kk + kbb + 8);
      a[mi] = join16(lo, hi);
    }
#pragma unroll
    for (int ni = 0; ni < 4; ++ni) {
      v16bf bfr = *(const v16bf*)(Brow[ni] + kk + kbb);
#pragma unroll
      for (int mi = 0; mi < 4; ++mi)
        acc[mi][ni] = __builtin_amdgcn_wmma_f32_16x16x32_bf16(
            false, a[mi], false, bfr, (short)0, acc[mi][ni], false, false);
    }
  }

  if (z < 2) {
    __bf16* dst = (z == 0) ? qout : kout;
#pragma unroll
    for (int ni = 0; ni < 4; ++ni) {
      const int col = n0 + ni * 16 + l15;
      const float bv_ = bias[col];
      const int h = col >> 7, d = col & 127;
#pragma unroll
      for (int mi = 0; mi < 4; ++mi) {
#pragma unroll
        for (int r = 0; r < 8; ++r) {
          const int row = m0 + mi * 16 + r + grp * 8;
          const int bI = row >> 11, s = row & (S_DIM - 1);
          dst[(((size_t)bI * H_DIM + h) * S_DIM + s) * D_DIM + d] =
              (__bf16)(acc[mi][ni][r] + bv_);
        }
      }
    }
  } else {
    // V^T: per lane, 8 consecutive s values of one (h,d) column -> one 16B store
#pragma unroll
    for (int ni = 0; ni < 4; ++ni) {
      const int col = n0 + ni * 16 + l15;
      const float bv_ = bias[col];
      const int h = col >> 7, d = col & 127;
#pragma unroll
      for (int mi = 0; mi < 4; ++mi) {
        const int rowb = m0 + mi * 16 + grp * 8;
        const int bI = rowb >> 11, sb = rowb & (S_DIM - 1);
        unsigned int pk[4];
#pragma unroll
        for (int r = 0; r < 4; ++r)
          pk[r] = (unsigned int)f2bf_bits(acc[mi][ni][2 * r] + bv_) |
                  ((unsigned int)f2bf_bits(acc[mi][ni][2 * r + 1] + bv_) << 16);
        uint4 val = {pk[0], pk[1], pk[2], pk[3]};
        *(uint4*)(vtout + (((size_t)bI * H_DIM + h) * D_DIM + d) * S_DIM + sb) = val;
      }
    }
  }
}

// ---------------------------------------------------------------------------
// RoPE on Q (always) and K (if *k_rope), in place on [B,H,S,D] bf16
// ---------------------------------------------------------------------------
__global__ __launch_bounds__(256) void rope_kernel(__bf16* __restrict__ Q,
                                                   __bf16* __restrict__ K,
                                                   const float* __restrict__ angles,
                                                   const int* __restrict__ k_rope) {
  const int t = blockIdx.x * blockDim.x + threadIdx.x;  // B*H*S*64 threads
  const int j = t & 63;
  const int s = (t >> 6) & (S_DIM - 1);
  const int bh = t >> 17;
  const float sn = angles[s * 64 + j];
  const float cs = angles[S_DIM * 64 + s * 64 + j];
  const size_t base = ((size_t)bh * S_DIM + s) * D_DIM;
  float q1 = (float)Q[base + j], q2 = (float)Q[base + 64 + j];
  Q[base + j] = (__bf16)(q1 * cs - q2 * sn);
  Q[base + 64 + j] = (__bf16)(q1 * sn + q2 * cs);
  if (k_rope[0]) {
    float k1 = (float)K[base + j], k2 = (float)K[base + 64 + j];
    K[base + j] = (__bf16)(k1 * cs - k2 * sn);
    K[base + 64 + j] = (__bf16)(k1 * sn + k2 * cs);
  }
}

// ---------------------------------------------------------------------------
// Flash attention: Q[B,H,S,D], K[B,H,S,D], V^T[B,H,D,S] -> O[B,S,H*D] (bf16)
// One wave owns 16 q-rows. K/V blocks (32 keys) are DMA'd into LDS once per
// workgroup with global_load_async_to_lds_b128, double-buffered so the DMA
// overlaps the 16 WMMAs of the current block. fp32 online softmax.
// ---------------------------------------------------------------------------
__global__ __launch_bounds__(256) void flash_attn(
    const __bf16* __restrict__ Q, const __bf16* __restrict__ K,
    const __bf16* __restrict__ Vt, const int* __restrict__ amask,
    const int* __restrict__ causal, __bf16* __restrict__ O) {
  __shared__ __align__(16) __bf16 Klds[2][32][136];   // [buf][key][d]  (+8 pad: bank stagger)
  __shared__ __align__(16) __bf16 Vlds[2][128][40];   // [buf][d][key]  (+8 pad)
  __shared__ __align__(32) __bf16 plds[8][16][40];    // P tile per wave

  const int tid = threadIdx.x;
  const int lane = tid & 31;
  const int wave = tid >> 5;
  const int bh = blockIdx.y;
  const int bI = bh >> 4, h = bh & 15;
  const int q0 = blockIdx.x * 128 + wave * 16;
  const int l15 = lane & 15;
  const int grp = (lane >= 16) ? 1 : 0;
  const int kb8 = grp * 8;
  const int kbb = grp * 16;
  const float scale = 0.08838834764831845f;  // 1/sqrt(128)
  const int cz = causal[0];

  const __bf16* Qb = Q + ((size_t)bh * S_DIM + q0) * D_DIM;
  const __bf16* Kb = K + (size_t)bh * S_DIM * D_DIM;
  const __bf16* Vb = Vt + (size_t)bh * D_DIM * S_DIM;
  const int* mrow = amask + bI * S_DIM;

  // cooperative async copy of one 32-key K block (8KB) + V^T block (8KB):
  // 512 16B chunks each, 2 per thread per tensor
  auto issue_kv = [&](int kbN, int bufN) {
#pragma unroll
    for (int c = 0; c < 2; ++c) {
      const int ck = tid + c * 256;
      const int kr = ck >> 4, kc = ck & 15;          // K: 32 rows x 16 chunks
      async_ld16(Kb + (size_t)(kbN + kr) * D_DIM + kc * 8, &Klds[bufN][kr][kc * 8]);
      const int vr = ck >> 2, vc = ck & 3;           // V: 128 rows x 4 chunks
      async_ld16(Vb + (size_t)vr * S_DIM + kbN + vc * 8, &Vlds[bufN][vr][vc * 8]);
    }
  };

  v16bf qf[4];
#pragma unroll
  for (int dc = 0; dc < 4; ++dc) {
    const __bf16* p = Qb + (size_t)l15 * D_DIM + dc * 32;
    qf[dc] = join16(*(const v8bf*)(p + kbb), *(const v8bf*)(p + kbb + 8));
  }

  v8f acc[8];
#pragma unroll
  for (int dt = 0; dt < 8; ++dt) acc[dt] = {};
  float mstat[8], lstat[8];
#pragma unroll
  for (int r = 0; r < 8; ++r) { mstat[r] = -3.0e38f; lstat[r] = 0.f; }

  int buf = 0;
  issue_kv(0, 0);

#pragma unroll 1
  for (int kb = 0; kb < S_DIM; kb += 32) {
    // wait for this wave's DMA chunks, then publish the buffer to all waves
    asm volatile("s_wait_asynccnt 0" ::: "memory");
    __syncthreads();
    if (kb + 32 < S_DIM) issue_kv(kb + 32, buf ^ 1);  // overlap next block's DMA

    v8f s0 = {}, s1 = {};
#pragma unroll
    for (int dc = 0; dc < 4; ++dc) {
      const int col = dc * 32 + kbb;
      v16bf k0 = join16(*(const v8bf*)&Klds[buf][l15][col],
                        *(const v8bf*)&Klds[buf][l15][col + 8]);
      s0 = __builtin_amdgcn_wmma_f32_16x16x32_bf16(false, qf[dc], false, k0,
                                                   (short)0, s0, false, false);
      v16bf k1 = join16(*(const v8bf*)&Klds[buf][16 + l15][col],
                        *(const v8bf*)&Klds[buf][16 + l15][col + 8]);
      s1 = __builtin_amdgcn_wmma_f32_16x16x32_bf16(false, qf[dc], false, k1,
                                                   (short)0, s1, false, false);
    }

    const int mv0 = mrow[kb + l15];
    const int mv1 = mrow[kb + 16 + l15];
    const int key0 = kb + l15, key1 = kb + 16 + l15;

#pragma unroll
    for (int r = 0; r < 8; ++r) {
      const int qr = q0 + r + grp * 8;
      float a0 = (mv0 == 0 || (cz && key0 > qr)) ? -1e30f : s0[r] * scale;
      float a1 = (mv1 == 0 || (cz && key1 > qr)) ? -1e30f : s1[r] * scale;
      float rm = fmaxf(a0, a1);
      rm = fmaxf(rm, __shfl_xor(rm, 8, 16));
      rm = fmaxf(rm, __shfl_xor(rm, 4, 16));
      rm = fmaxf(rm, __shfl_xor(rm, 2, 16));
      rm = fmaxf(rm, __shfl_xor(rm, 1, 16));
      const float mn = fmaxf(mstat[r], rm);
      const float alpha = __expf(mstat[r] - mn);
      const float p0 = __expf(a0 - mn);
      const float p1 = __expf(a1 - mn);
      float rs = p0 + p1;
      rs += __shfl_xor(rs, 8, 16);
      rs += __shfl_xor(rs, 4, 16);
      rs += __shfl_xor(rs, 2, 16);
      rs += __shfl_xor(rs, 1, 16);
      lstat[r] = lstat[r] * alpha + rs;
      mstat[r] = mn;
#pragma unroll
      for (int dt = 0; dt < 8; ++dt) acc[dt][r] *= alpha;
      plds[wave][r + grp * 8][l15] = (__bf16)p0;
      plds[wave][r + grp * 8][16 + l15] = (__bf16)p1;
    }

    // per-wave LDS round-trip to convert P from C-layout to A-layout
    asm volatile("s_wait_dscnt 0" ::: "memory");
    v16bf pA = join16(*(const v8bf*)&plds[wave][l15][kbb],
                      *(const v8bf*)&plds[wave][l15][kbb + 8]);

#pragma unroll
    for (int dt = 0; dt < 8; ++dt) {
      v16bf vf = join16(*(const v8bf*)&Vlds[buf][dt * 16 + l15][kbb],
                        *(const v8bf*)&Vlds[buf][dt * 16 + l15][kbb + 8]);
      acc[dt] = __builtin_amdgcn_wmma_f32_16x16x32_bf16(false, pA, false, vf,
                                                        (short)0, acc[dt], false, false);
    }
    buf ^= 1;
  }

#pragma unroll
  for (int r = 0; r < 8; ++r) {
    const float inv = (lstat[r] > 0.f) ? 1.f / lstat[r] : 0.f;
    const size_t row = (size_t)bI * S_DIM + q0 + r + grp * 8;
#pragma unroll
    for (int dt = 0; dt < 8; ++dt)
      O[row * E_DIM + h * D_DIM + dt * 16 + l15] = (__bf16)(acc[dt][r] * inv);
  }
}

// ---------------------------------------------------------------------------
// Output projection: out = A @ Wo^T + bo (fp32 out), same WMMA skeleton
// ---------------------------------------------------------------------------
__global__ __launch_bounds__(256) void out_gemm(const __bf16* __restrict__ A,
                                                const __bf16* __restrict__ W,
                                                const float* __restrict__ bias,
                                                float* __restrict__ out) {
  const int lane = threadIdx.x & 31;
  const int wave = threadIdx.x >> 5;
  const int m0 = blockIdx.x * 128 + (wave & 1) * 64;
  const int n0 = blockIdx.y * 256 + (wave >> 1) * 64;
  const int l15 = lane & 15;
  const int grp = (lane >= 16) ? 1 : 0;
  const int kb8 = grp * 8;
  const int kbb = grp * 16;

  v8f acc[4][4];
#pragma unroll
  for (int mi = 0; mi < 4; ++mi)
#pragma unroll
    for (int ni = 0; ni < 4; ++ni) acc[mi][ni] = {};

  const __bf16* Arow[4];
  const __bf16* Brow[4];
#pragma unroll
  for (int i = 0; i < 4; ++i) {
    Arow[i] = A + (size_t)(m0 + i * 16 + l15) * E_DIM;
    Brow[i] = W + (size_t)(n0 + i * 16 + l15) * E_DIM;
  }

#pragma unroll 2
  for (int kk = 0; kk < E_DIM; kk += 32) {
    v16bf a[4];
#pragma unroll
    for (int mi = 0; mi < 4; ++mi)
      a[mi] = join16(*(const v8bf*)(Arow[mi] + kk + kbb),
                     *(const v8bf*)(Arow[mi] + kk + kbb + 8));
#pragma unroll
    for (int ni = 0; ni < 4; ++ni) {
      v16bf bfr = *(const v16bf*)(Brow[ni] + kk + kbb);
#pragma unroll
      for (int mi = 0; mi < 4; ++mi)
        acc[mi][ni] = __builtin_amdgcn_wmma_f32_16x16x32_bf16(
            false, a[mi], false, bfr, (short)0, acc[mi][ni], false, false);
    }
  }

#pragma unroll
  for (int ni = 0; ni < 4; ++ni) {
    const int col = n0 + ni * 16 + l15;
    const float bv_ = bias[col];
#pragma unroll
    for (int mi = 0; mi < 4; ++mi)
#pragma unroll
      for (int r = 0; r < 8; ++r) {
        const int row = m0 + mi * 16 + r + grp * 8;
        out[(size_t)row * E_DIM + col] = acc[mi][ni][r] + bv_;
      }
  }
}

// ---------------------------------------------------------------------------
extern "C" void kernel_launch(void* const* d_in, const int* in_sizes, int n_in,
                              void* d_out, int out_size, void* d_ws, size_t ws_size,
                              hipStream_t stream) {
  (void)in_sizes; (void)n_in; (void)out_size; (void)ws_size;
  const float* x_q   = (const float*)d_in[0];
  const float* x_kv  = (const float*)d_in[1];
  const float* Wq    = (const float*)d_in[2];
  const float* bq    = (const float*)d_in[3];
  const float* Wk    = (const float*)d_in[4];
  const float* bk    = (const float*)d_in[5];
  const float* Wv    = (const float*)d_in[6];
  const float* bv    = (const float*)d_in[7];
  const float* Wo    = (const float*)d_in[8];
  const float* bo    = (const float*)d_in[9];
  const float* angles = (const float*)d_in[10];
  const int* attn_mask = (const int*)d_in[11];
  const int* causal  = (const int*)d_in[12];
  const int* k_rope  = (const int*)d_in[13];
  float* out = (float*)d_out;

  char* ws = (char*)d_ws;
  const size_t MB = 1024ull * 1024ull;
  __bf16* xq_b  = (__bf16*)(ws + 0 * MB);    // 32MB
  __bf16* xkv_b = (__bf16*)(ws + 32 * MB);   // 32MB
  __bf16* wq_b  = (__bf16*)(ws + 64 * MB);   // 8MB
  __bf16* wk_b  = (__bf16*)(ws + 72 * MB);   // 8MB
  __bf16* wv_b  = (__bf16*)(ws + 80 * MB);   // 8MB
  __bf16* wo_b  = (__bf16*)(ws + 88 * MB);   // 8MB
  __bf16* Qb    = (__bf16*)(ws + 96 * MB);   // 32MB  [B,H,S,D]
  __bf16* Kb    = (__bf16*)(ws + 128 * MB);  // 32MB  [B,H,S,D]
  __bf16* Vt    = (__bf16*)(ws + 160 * MB);  // 32MB  [B,H,D,S]
  __bf16* At    = (__bf16*)(ws + 192 * MB);  // 32MB  [B,S,E]

  const int nX4 = (B_DIM * S_DIM * E_DIM) / 4;  // 4194304
  const int nW4 = (E_DIM * E_DIM) / 4;          // 1048576
  cvt_f32_bf16<<<nX4 / 256, 256, 0, stream>>>(x_q, xq_b, nX4);
  cvt_f32_bf16<<<nX4 / 256, 256, 0, stream>>>(x_kv, xkv_b, nX4);
  cvt_f32_bf16<<<nW4 / 256, 256, 0, stream>>>(Wq, wq_b, nW4);
  cvt_f32_bf16<<<nW4 / 256, 256, 0, stream>>>(Wk, wk_b, nW4);
  cvt_f32_bf16<<<nW4 / 256, 256, 0, stream>>>(Wv, wv_b, nW4);
  cvt_f32_bf16<<<nW4 / 256, 256, 0, stream>>>(Wo, wo_b, nW4);

  qkv_gemm<<<dim3(64, 8, 3), 256, 0, stream>>>(xq_b, xkv_b, wq_b, wk_b, wv_b,
                                               bq, bk, bv, Qb, Kb, Vt);

  rope_kernel<<<(B_DIM * H_DIM * S_DIM * 64) / 256, 256, 0, stream>>>(Qb, Kb, angles, k_rope);

  flash_attn<<<dim3(16, B_DIM * H_DIM), 256, 0, stream>>>(Qb, Kb, Vt, attn_mask, causal, At);

  out_gemm<<<dim3(64, 8), 256, 0, stream>>>(At, wo_b, bo, out);
}
